// MultiHeadAttn_82102594830785
// MI455X (gfx1250) — matrix operand order repaired and checked by reading, not compiled, not run on hardware
//
#include <hip/hip_runtime.h>

// Problem constants (fixed by the reference)
#define Bn   2
#define Sn   2048
#define Dn   2048
#define Hn   16
#define HDn  128
#define Mn   (Bn * Sn)   // 4096 rows for the projection GEMMs
#define KB   64          // attention key-block size

typedef __attribute__((ext_vector_type(16))) __bf16 v16bf;
typedef __attribute__((ext_vector_type(8)))  __bf16 v8bf;
typedef __attribute__((ext_vector_type(4)))  __bf16 v4bf;
typedef __attribute__((ext_vector_type(8)))  float  v8f;

#define OUT_SPLIT     0   // bf16 out, [B,H,S,HD]
#define OUT_SPLIT_T   1   // bf16 out, [B,H,HD,S]  (V transposed for PV B-fragments)
#define OUT_PLAIN_F32 2   // f32 out,  [M,N]       (final projection -> d_out)

static __device__ __forceinline__ v8f wmma_bf16(v16bf a, v16bf b, v8f c) {
  // D = A(16x32) * B(32x16) + C, fp32 accumulate
  return __builtin_amdgcn_wmma_f32_16x16x32_bf16(false, a, false, b, (short)0, c,
                                                 false, false);
}

// ---- CDNA5 async Global->LDS (ASYNCcnt-tracked), per CDNA5 ISA ----
static __device__ __forceinline__ void async_ld_b128(void* lds_dst, const void* gsrc) {
  unsigned int lds_off = (unsigned int)(unsigned long long)lds_dst; // addr[31:0] = LDS offset
  unsigned long long ga = (unsigned long long)gsrc;
  asm volatile("global_load_async_to_lds_b128 %0, %1, off"
               :: "v"(lds_off), "v"(ga)
               : "memory");
}
static __device__ __forceinline__ void wait_asynccnt0() {
  asm volatile("s_wait_asynccnt 0x0" ::: "memory");
}

// A-fragment (16x32 bf16, row-major source, lda in elements).
// Lane l: row m, elems 0..15 = K (hi?16:0)+i (contiguous).
static __device__ __forceinline__ v16bf load_a_frag(const __bf16* base, int lda,
                                                    int m, int k0, int lane) {
  const int hi = (lane >> 4) & 1;
  const __bf16* row = base + (size_t)m * lda + k0 + hi * 16;
  v8bf lo = *(const v8bf*)(row);
  v8bf hh = *(const v8bf*)(row + 8);
  v16bf a;
#pragma unroll
  for (int i = 0; i < 8; ++i) { a[i] = lo[i]; a[8 + i] = hh[i]; }
  return a;
}

// B-fragment (32x16 bf16): lane holds col n = l%16, 16 contiguous K (hi -> K 16..31).
static __device__ __forceinline__ v16bf load_b_frag(const __bf16* base, int ldb,
                                                    int n, int k0, int lane) {
  const int hi = (lane >> 4) & 1;
  return *(const v16bf*)(base + (size_t)n * ldb + k0 + hi * 16);
}

// ---------------------------------------------------------------- fp32 -> bf16
__global__ __launch_bounds__(256)
void mha_cvt_bf16(const float* __restrict__ in, __bf16* __restrict__ out, int n4) {
  int i = blockIdx.x * blockDim.x + threadIdx.x;
  if (i < n4) {
    float4 f = ((const float4*)in)[i];
    v4bf o;
    o[0] = (__bf16)f.x; o[1] = (__bf16)f.y; o[2] = (__bf16)f.z; o[3] = (__bf16)f.w;
    ((v4bf*)out)[i] = o;
  }
}

// ------------------------------------------------------------------- GEMM
// y[M,N] = A[M,K] @ W[N,K]^T + bias.  Block = 4 waves; each wave a 32x64 tile
// (2 A-frags x 4 B-frags -> 8 WMMA per 32-K chunk, B-frags reused 2x).
// grid = (N/64, M/128).  K = Dn = 2048.
template <int MODE>
__global__ __launch_bounds__(128)
void mha_gemm_bf16(const __bf16* __restrict__ A, const __bf16* __restrict__ W,
                   const float* __restrict__ bias, void* __restrict__ outp) {
  const int lane = threadIdx.x & 31;
  const int wv   = threadIdx.x >> 5;
  const int hi   = lane >> 4;
  const int nn   = lane & 15;
  const int n0   = blockIdx.x * 64;
  const int m0   = blockIdx.y * 128 + wv * 32;

  v8f acc[2][4];
#pragma unroll
  for (int mi = 0; mi < 2; ++mi)
#pragma unroll
    for (int j = 0; j < 4; ++j)
#pragma unroll
      for (int e = 0; e < 8; ++e) acc[mi][j][e] = 0.0f;

  for (int k0 = 0; k0 < Dn; k0 += 32) {
    if (k0 + 64 <= Dn) {  // global_prefetch_b8 of next A chunk
      __builtin_prefetch(A + (size_t)(m0 + nn) * Dn + k0 + 32, 0, 0);
    }
    v16bf a0 = load_a_frag(A, Dn, m0 + nn, k0, lane);
    v16bf a1 = load_a_frag(A, Dn, m0 + 16 + nn, k0, lane);
#pragma unroll
    for (int j = 0; j < 4; ++j) {
      v16bf bf = load_b_frag(W, Dn, n0 + j * 16 + nn, k0, lane);
      acc[0][j] = wmma_bf16(a0, bf, acc[0][j]);
      acc[1][j] = wmma_bf16(a1, bf, acc[1][j]);
    }
  }

#pragma unroll
  for (int mi = 0; mi < 2; ++mi) {
#pragma unroll
    for (int j = 0; j < 4; ++j) {
      const int n = n0 + j * 16 + nn;
      const float bv = bias[n];
#pragma unroll
      for (int r = 0; r < 8; ++r) {
        const int m = m0 + mi * 16 + r + hi * 8;
        const float v = acc[mi][j][r] + bv;
        if (MODE == OUT_PLAIN_F32) {
          ((float*)outp)[(size_t)m * Dn + n] = v;
        } else {
          const int b  = m >> 11;        // m / Sn
          const int s  = m & (Sn - 1);
          const int h  = n >> 7;         // n / HDn
          const int hd = n & (HDn - 1);
          size_t off;
          if (MODE == OUT_SPLIT)
            off = (((size_t)(b * Hn + h)) * Sn + s) * HDn + hd;
          else
            off = (((size_t)(b * Hn + h)) * HDn + hd) * Sn + s;
          ((__bf16*)outp)[off] = (__bf16)v;
        }
      }
    }
  }
}

// -------------------------------------------------------------- flash attention
// grid = (Sn/64, Hn, Bn), block = 128 (4 waves). Wave w owns q-rows
// [blockIdx.x*64 + w*16, +16). Key blocks of 64, double-buffered via async
// Global->LDS loads (ASYNCcnt) so staging of block i+1 overlaps compute of i.
__global__ __launch_bounds__(128)
void mha_flash_attn(const __bf16* __restrict__ Q, const __bf16* __restrict__ K,
                    const __bf16* __restrict__ Vt, __bf16* __restrict__ Oa) {
  __shared__ __bf16 Kbuf[2][KB * HDn];    // [key 0..63][hd 0..127]   2 x 16KB
  __shared__ __bf16 Vbuf[2][HDn * KB];    // [hd 0..127][key 0..63]   2 x 16KB
  __shared__ __bf16 Pt[4][16 * KB];       // per-wave probs [m][key]      8KB

  const int t    = threadIdx.x;
  const int lane = t & 31;
  const int wv   = t >> 5;
  const int hi   = lane >> 4;
  const int nn   = lane & 15;
  const int b = blockIdx.z, h = blockIdx.y;
  const int q0 = blockIdx.x * 64 + wv * 16;

  const __bf16* Qh = Q  + (((size_t)b * Hn + h) * Sn) * HDn;
  const __bf16* Kh = K  + (((size_t)b * Hn + h) * Sn) * HDn;
  const __bf16* Vh = Vt + (((size_t)b * Hn + h) * HDn) * Sn;

  // Staging maps (128 threads): K 64 rows x 128 cols -> 2 thr/row, 32 cols each from col 0/64;
  // Vt 128 rows x 64 cols -> 1 thr/row.
  const int krow = t >> 1, kcol = (t & 1) * 64;

  // Q fragments resident in registers: 4 K-chunks of 32 over HD=128
  v16bf qf[4];
#pragma unroll
  for (int c = 0; c < 4; ++c) qf[c] = load_a_frag(Qh, HDn, q0 + nn, c * 32, lane);

  v8f oacc[8];
#pragma unroll
  for (int tt = 0; tt < 8; ++tt)
#pragma unroll
    for (int e = 0; e < 8; ++e) oacc[tt][e] = 0.0f;
  float rmax[8], rsum[8];
#pragma unroll
  for (int r = 0; r < 8; ++r) { rmax[r] = -3.0e30f; rsum[r] = 0.0f; }

  const float scale = 0.08838834764831845f;  // 1/sqrt(128)
  const int NB = Sn / KB;

  // ---- prologue: async-stage block 0 into buffer 0
  {
    const __bf16* ks = Kh + (size_t)krow * HDn + kcol;
    const __bf16* vs = Vh + (size_t)t * Sn;
#pragma unroll
    for (int i = 0; i < 4; ++i) {
      async_ld_b128(&Kbuf[0][krow * HDn + kcol + 8 * i], ks + 8 * i);
      async_ld_b128(&Vbuf[0][t * KB + 8 * i], vs + 8 * i);
    }
  }

  for (int ib = 0; ib < NB; ++ib) {
    const int cur = ib & 1;
    wait_asynccnt0();   // my async writes into buf[cur] have landed in LDS
    __syncthreads();    // everyone's have; prev compute on buf[cur^1] done

    // ---- issue async staging of next block into the other buffer
    if (ib + 1 < NB) {
      const int kb2 = (ib + 1) * KB;
      const __bf16* ks = Kh + (size_t)(kb2 + krow) * HDn + kcol;
      const __bf16* vs = Vh + (size_t)t * Sn + kb2;
#pragma unroll
      for (int i = 0; i < 4; ++i) {
        async_ld_b128(&Kbuf[cur ^ 1][krow * HDn + kcol + 8 * i], ks + 8 * i);
        async_ld_b128(&Vbuf[cur ^ 1][t * KB + 8 * i], vs + 8 * i);
      }
    }

    const __bf16* Kt  = &Kbuf[cur][0];
    const __bf16* Vts = &Vbuf[cur][0];

    // ---- scores: S(16x64) = Q(16x128) * K^T : 4 n-tiles x 4 hd-chunks
    v8f sc[4];
#pragma unroll
    for (int nt = 0; nt < 4; ++nt) {
#pragma unroll
      for (int e = 0; e < 8; ++e) sc[nt][e] = 0.0f;
#pragma unroll
      for (int c = 0; c < 4; ++c) {
        v16bf bf = *(const v16bf*)(&Kt[(nt * 16 + nn) * HDn + c * 32 + hi * 16]);
        sc[nt] = wmma_bf16(qf[c], bf, sc[nt]);
      }
    }

    // ---- online softmax (row = r + hi*8; 16-lane horizontal reductions)
#pragma unroll
    for (int r = 0; r < 8; ++r) {
      float s0 = sc[0][r] * scale, s1 = sc[1][r] * scale;
      float s2 = sc[2][r] * scale, s3 = sc[3][r] * scale;
      float mx = fmaxf(fmaxf(s0, s1), fmaxf(s2, s3));
      mx = fmaxf(mx, __shfl_xor(mx, 1, 32));
      mx = fmaxf(mx, __shfl_xor(mx, 2, 32));
      mx = fmaxf(mx, __shfl_xor(mx, 4, 32));
      mx = fmaxf(mx, __shfl_xor(mx, 8, 32));
      const float mnew = fmaxf(rmax[r], mx);
      const float fac  = __expf(rmax[r] - mnew);
      const float p0 = __expf(s0 - mnew), p1 = __expf(s1 - mnew);
      const float p2 = __expf(s2 - mnew), p3 = __expf(s3 - mnew);
      float ps = (p0 + p1) + (p2 + p3);
      ps += __shfl_xor(ps, 1, 32);
      ps += __shfl_xor(ps, 2, 32);
      ps += __shfl_xor(ps, 4, 32);
      ps += __shfl_xor(ps, 8, 32);
      rsum[r] = rsum[r] * fac + ps;
      rmax[r] = mnew;
#pragma unroll
      for (int tt = 0; tt < 8; ++tt) oacc[tt][r] *= fac;
      __bf16* pr = &Pt[wv][(r + hi * 8) * KB + nn];
      pr[0]  = (__bf16)p0;
      pr[16] = (__bf16)p1;
      pr[32] = (__bf16)p2;
      pr[48] = (__bf16)p3;
    }

    // ---- PV: O(16x128) += P(16x64) * V(64x128); 2 key-chunks x 8 hd-tiles
    v16bf pa0 = load_a_frag(&Pt[wv][0], KB, nn, 0, lane);
    v16bf pa1 = load_a_frag(&Pt[wv][0], KB, nn, 32, lane);
#pragma unroll
    for (int tt = 0; tt < 8; ++tt) {
      v16bf vb0 = *(const v16bf*)(&Vts[(tt * 16 + nn) * KB + hi * 16]);
      v16bf vb1 = *(const v16bf*)(&Vts[(tt * 16 + nn) * KB + 32 + hi * 16]);
      oacc[tt] = wmma_bf16(pa0, vb0, oacc[tt]);
      oacc[tt] = wmma_bf16(pa1, vb1, oacc[tt]);
    }
    // next iteration's barrier separates these reads from buffer reuse
  }

  // ---- epilogue: normalize and write merged [B,S,D] bf16
#pragma unroll
  for (int tt = 0; tt < 8; ++tt) {
#pragma unroll
    for (int r = 0; r < 8; ++r) {
      const int srow = q0 + r + hi * 8;
      const float v = oacc[tt][r] / rsum[r];
      Oa[((size_t)b * Sn + srow) * Dn + h * HDn + tt * 16 + nn] = (__bf16)v;
    }
  }
}

// ------------------------------------------------------------------- launch
extern "C" void kernel_launch(void* const* d_in, const int* in_sizes, int n_in,
                              void* d_out, int out_size, void* d_ws, size_t ws_size,
                              hipStream_t stream) {
  const float* x  = (const float*)d_in[0];
  const float* Wq = (const float*)d_in[1];
  const float* bq = (const float*)d_in[2];
  const float* Wk = (const float*)d_in[3];
  const float* bk = (const float*)d_in[4];
  const float* Wv = (const float*)d_in[5];
  const float* bv = (const float*)d_in[6];
  const float* Wo = (const float*)d_in[7];
  const float* bo = (const float*)d_in[8];

  // Workspace carve (bf16 buffers, 256B aligned). Total ~118 MB.
  char* p = (char*)d_ws;
  auto carve = [&](size_t elems) {
    __bf16* r = (__bf16*)p;
    p += ((elems * sizeof(__bf16) + 255) & ~(size_t)255);
    return r;
  };
  __bf16* xb   = carve((size_t)Mn * Dn);        // x in bf16        [M,D]
  __bf16* wqb  = carve((size_t)Dn * Dn);
  __bf16* wkb  = carve((size_t)Dn * Dn);
  __bf16* wvb  = carve((size_t)Dn * Dn);
  __bf16* wob  = carve((size_t)Dn * Dn);
  __bf16* qb   = carve((size_t)Mn * Dn);        // Q  [B,H,S,HD]
  __bf16* kb_  = carve((size_t)Mn * Dn);        // K  [B,H,S,HD]
  __bf16* vtb  = carve((size_t)Mn * Dn);        // Vt [B,H,HD,S]
  __bf16* attn = carve((size_t)Mn * Dn);        // merged attn [B,S,D]

  // 1) fp32 -> bf16 conversions
  {
    int n4 = Mn * Dn / 4;
    mha_cvt_bf16<<<n4 / 256, 256, 0, stream>>>(x, xb, n4);
    int w4 = Dn * Dn / 4;
    mha_cvt_bf16<<<w4 / 256, 256, 0, stream>>>(Wq, wqb, w4);
    mha_cvt_bf16<<<w4 / 256, 256, 0, stream>>>(Wk, wkb, w4);
    mha_cvt_bf16<<<w4 / 256, 256, 0, stream>>>(Wv, wvb, w4);
    mha_cvt_bf16<<<w4 / 256, 256, 0, stream>>>(Wo, wob, w4);
  }

  // 2) QKV projections (Q/K head-split, V head-split + transposed)
  dim3 gg(Dn / 64, Mn / 128);
  mha_gemm_bf16<OUT_SPLIT><<<gg, 128, 0, stream>>>(xb, wqb, bq, qb);
  mha_gemm_bf16<OUT_SPLIT><<<gg, 128, 0, stream>>>(xb, wkb, bk, kb_);
  mha_gemm_bf16<OUT_SPLIT_T><<<gg, 128, 0, stream>>>(xb, wvb, bv, vtb);

  // 3) flash attention (double-buffered async Global->LDS staging)
  dim3 ga(Sn / 64, Hn, Bn);
  mha_flash_attn<<<ga, 128, 0, stream>>>(qb, kb_, vtb, attn);

  // 4) output projection -> fp32 d_out
  mha_gemm_bf16<OUT_PLAIN_F32><<<gg, 128, 0, stream>>>(attn, wob, bo, d_out);

  (void)in_sizes; (void)n_in; (void)out_size; (void)ws_size;
}
